// RQVAE_90031104459320
// MI455X (gfx1250) — compile-verified, matrix operand order checked by reading, not run on hardware
//
#include <hip/hip_runtime.h>
#include <hip/hip_bf16.h>

typedef __bf16 bf16;
typedef __attribute__((ext_vector_type(8)))  __bf16 v8bf;
typedef __attribute__((ext_vector_type(16))) __bf16 v16bf;
typedef __attribute__((ext_vector_type(2)))  float  v2f;
typedef __attribute__((ext_vector_type(4)))  float  v4f;
typedef __attribute__((ext_vector_type(8)))  float  v8f;

#define N_ROWS 131072
#define RQ_BETA 0.25f
#define CB_PAD 34   // LDS row stride (dwords) for the staged codebook level

// -------------------- small utility kernels --------------------
__global__ void k_zero(float* acc) {
    if (threadIdx.x < 4) acc[threadIdx.x] = 0.f;
}

__global__ void k_finalize(const float* __restrict__ acc, float* __restrict__ out) {
    float s = 0.f;
    for (int l = 0; l < 4; ++l)
        s += (1.f + RQ_BETA) * acc[l] / ((float)N_ROWS * 32.f);
    out[0] = s * 0.25f;
}

// -------------------- weight repack: W (KxN f32, row-major) -> bf16 B-fragment order
// Fragment f = nt*KT + kt; within fragment: [lane 0..31][half 0..15] contiguous.
// B element (k, n): n = nt*16 + (lane&15); k = kt*32 + h + (h>=8)*8 + (lane>=16)*8.
__global__ void k_repack(const float* __restrict__ w, bf16* __restrict__ dst, int K, int N) {
    int tid = blockIdx.x * blockDim.x + threadIdx.x;
    int total = K * N;
    if (tid >= total) return;
    int KT = K >> 5;
    int h  = tid & 15;
    int l  = (tid >> 4) & 31;
    int f  = tid >> 9;
    int kt = f % KT;
    int nt = f / KT;
    int n  = (nt << 4) + (l & 15);
    int k  = (kt << 5) + h + ((h >= 8) ? 8 : 0) + ((l >= 16) ? 8 : 0);
    dst[tid] = (bf16)w[(size_t)k * N + n];
}

// -------------------- A-fragment loaders (ISA 16-bit A 16x32 layout) --------------------
__device__ inline v16bf load_a_bf16(const bf16* __restrict__ p) {
    union { v16bf v; v8bf h[2]; } u;
    u.h[0] = *(const v8bf*)(p);
    u.h[1] = *(const v8bf*)(p + 16);
    return u.v;
}
__device__ inline v16bf load_a_f32(const float* __restrict__ p) {
    union { v16bf v; v8bf h[2]; } u;
#pragma unroll
    for (int i = 0; i < 8; ++i) u.h[0][i] = (bf16)p[i];
#pragma unroll
    for (int i = 0; i < 8; ++i) u.h[1][i] = (bf16)p[16 + i];
    return u.v;
}

// -------------------- bf16 WMMA GEMM: C(MxN) = act(MxK) @ W(KxN) + bias --------------------
// Block: 256 threads = 8 waves; block tile 128 rows x 64 cols; wave tile 32x32 (2x2 frags).
// Compile-time shapes; explicit 2-stage pipeline: prefetch kt+1 while kt's WMMAs issue.
template <int K, int N, bool AF32, bool CF32, bool RELU>
__global__ __launch_bounds__(256)
void k_gemm(const void* __restrict__ Ain,
            const bf16* __restrict__ Bfrag, const float* __restrict__ bias,
            void* __restrict__ Cout) {
    constexpr int KT = K >> 5;
    constexpr int NT = N >> 4;
    const int lane = threadIdx.x & 31;
    const int wave = threadIdx.x >> 5;
    const int wm = wave & 3;
    const int wn = wave >> 2;
    if (NT < 4 && wn != 0) return;   // N==32: only wn==0 waves have work (wave-uniform exit)

    const int mBase = blockIdx.y * 128 + wm * 32;
    const int nt0 = blockIdx.x * 4 + wn * 2;
    const int nt1 = nt0 + 1;
    const int ml = lane & 15;
    const int kboff = (lane >> 4) << 3;

    const float* a0f = (const float*)Ain + (size_t)(mBase + ml) * K + kboff;
    const float* a1f = (const float*)Ain + (size_t)(mBase + 16 + ml) * K + kboff;
    const bf16*  a0b = (const bf16*)Ain + (size_t)(mBase + ml) * K + kboff;
    const bf16*  a1b = (const bf16*)Ain + (size_t)(mBase + 16 + ml) * K + kboff;
    const bf16*  b0p = Bfrag + (((size_t)nt0 * KT) * 32 + lane) * 16;
    const bf16*  b1p = Bfrag + (((size_t)nt1 * KT) * 32 + lane) * 16;

    v8f c00 = {}, c01 = {}, c10 = {}, c11 = {};

    v16bf A0, A1, B0, B1;
    A0 = AF32 ? load_a_f32(a0f) : load_a_bf16(a0b);
    A1 = AF32 ? load_a_f32(a1f) : load_a_bf16(a1b);
    B0 = *(const v16bf*)(b0p);
    B1 = *(const v16bf*)(b1p);

#pragma unroll
    for (int kt = 0; kt < KT; ++kt) {
        v16bf NA0, NA1, NB0, NB1;
        if (kt + 1 < KT) {
            NA0 = AF32 ? load_a_f32(a0f + (kt + 1) * 32) : load_a_bf16(a0b + (kt + 1) * 32);
            NA1 = AF32 ? load_a_f32(a1f + (kt + 1) * 32) : load_a_bf16(a1b + (kt + 1) * 32);
            NB0 = *(const v16bf*)(b0p + (kt + 1) * 512);
            NB1 = *(const v16bf*)(b1p + (kt + 1) * 512);
        }
        c00 = __builtin_amdgcn_wmma_f32_16x16x32_bf16(false, A0, false, B0, (short)0, c00, false, false);
        c10 = __builtin_amdgcn_wmma_f32_16x16x32_bf16(false, A1, false, B0, (short)0, c10, false, false);
        c01 = __builtin_amdgcn_wmma_f32_16x16x32_bf16(false, A0, false, B1, (short)0, c01, false, false);
        c11 = __builtin_amdgcn_wmma_f32_16x16x32_bf16(false, A1, false, B1, (short)0, c11, false, false);
        if (kt + 1 < KT) { A0 = NA0; A1 = NA1; B0 = NB0; B1 = NB1; }
    }

    // Epilogue: C-frag layout: vgpr r -> M = r + 8*(lane>=16); N = lane&15 (+16*nt)
    const int mh = (lane >> 4) << 3;
    const float bb0 = bias[(nt0 << 4) + ml];
    const float bb1 = bias[(nt1 << 4) + ml];
#pragma unroll
    for (int r = 0; r < 8; ++r) {
        const int m0 = mBase + mh + r;
        const int m1 = m0 + 16;
        float v00 = c00[r] + bb0, v01 = c01[r] + bb1;
        float v10 = c10[r] + bb0, v11 = c11[r] + bb1;
        if (RELU) {
            v00 = fmaxf(v00, 0.f); v01 = fmaxf(v01, 0.f);
            v10 = fmaxf(v10, 0.f); v11 = fmaxf(v11, 0.f);
        }
        const int n0 = (nt0 << 4) + ml, n1 = (nt1 << 4) + ml;
        if (CF32) {
            float* C = (float*)Cout;
            C[(size_t)m0 * N + n0] = v00; C[(size_t)m0 * N + n1] = v01;
            C[(size_t)m1 * N + n0] = v10; C[(size_t)m1 * N + n1] = v11;
        } else {
            bf16* C = (bf16*)Cout;
            C[(size_t)m0 * N + n0] = (bf16)v00; C[(size_t)m0 * N + n1] = (bf16)v01;
            C[(size_t)m1 * N + n0] = (bf16)v10; C[(size_t)m1 * N + n1] = (bf16)v11;
        }
    }
}

// -------------------- residual VQ: exact-f32 scoring via v_wmma_f32_16x16x4_f32 ----------
// Block: 256 threads = 8 waves, each wave one 16-row tile (block covers 128 rows).
// Codebook level staged in LDS (padded stride for bank spread); B-frags from ds_load_b64.
// argmin over 256 codes: d = ||e||^2 - 2*res.e (||res||^2 dropped: constant per row).
__global__ __launch_bounds__(256)
void k_vq(const float* __restrict__ Z, const float* __restrict__ cb,
          bf16* __restrict__ XQ, float* __restrict__ idx_out, float* __restrict__ acc) {
    __shared__ float s_cb[256 * CB_PAD];
    __shared__ float s_en[256];
    __shared__ int   s_idx[8][16];
    const int lane = threadIdx.x & 31;
    const int wave = threadIdx.x >> 5;
    const int ml   = lane & 15;
    const int hi   = lane >> 4;
    const int rowBase = blockIdx.x * 128 + wave * 16;
    const int row     = rowBase + ml;

    // res held in f32-WMMA A-frag layout: lane(k-slot) k = kt*4 + 2*hi + s, kt<8, s<2
    float res[16], z0[16];
    {
        const float* zp = Z + (size_t)row * 32;
#pragma unroll
        for (int kt = 0; kt < 8; ++kt) {
            res[2 * kt]     = zp[kt * 4 + 2 * hi];
            res[2 * kt + 1] = zp[kt * 4 + 2 * hi + 1];
        }
#pragma unroll
        for (int i = 0; i < 16; ++i) z0[i] = res[i];
    }

    float lacc[4];
    for (int l = 0; l < 4; ++l) {
        // stage codebook level into LDS (one code row per thread) + norms on the fly
        {
            const float* ec = cb + ((size_t)l * 256 + threadIdx.x) * 32;
            float* dst = s_cb + threadIdx.x * CB_PAD;
            float s = 0.f;
#pragma unroll
            for (int q = 0; q < 8; ++q) {
                v4f v = *(const v4f*)(ec + q * 4);
                s += v[0] * v[0] + v[1] * v[1] + v[2] * v[2] + v[3] * v[3];
                *(v2f*)(dst + q * 4)     = v2f{ v[0], v[1] };
                *(v2f*)(dst + q * 4 + 2) = v2f{ v[2], v[3] };
            }
            s_en[threadIdx.x] = s;
        }
        __syncthreads();

        float bd[8]; int bn[8];
#pragma unroll
        for (int r = 0; r < 8; ++r) { bd[r] = 3.4e38f; bn[r] = 0; }

        for (int nt = 0; nt < 16; ++nt) {
            const int n = nt * 16 + ml;
            const float* en = s_cb + n * CB_PAD + 2 * hi;
            // batch all 8 B-fragment LDS loads, then the WMMA chain
            v2f bfr[8];
#pragma unroll
            for (int kt = 0; kt < 8; ++kt) bfr[kt] = *(const v2f*)(en + kt * 4);
            v8f sc = {};
#pragma unroll
            for (int kt = 0; kt < 8; ++kt) {
                v2f a = { res[2 * kt], res[2 * kt + 1] };
                sc = __builtin_amdgcn_wmma_f32_16x16x4_f32(false, a, false, bfr[kt], (short)0, sc, false, false);
            }
            const float enm = s_en[n];
#pragma unroll
            for (int r = 0; r < 8; ++r) {
                float d = enm - 2.f * sc[r];
                if (d < bd[r]) { bd[r] = d; bn[r] = n; }
            }
        }
        // butterfly min-reduce across the 16 lanes of each half (C-frag N lives in lane&15)
        // NOTE: non-short-circuit logic -> pure cmp/cndmask, no EXEC divergence
#pragma unroll
        for (int r = 0; r < 8; ++r) {
#pragma unroll
            for (int off = 1; off < 16; off <<= 1) {
                float od = __shfl_xor(bd[r], off, 32);
                int   on = __shfl_xor(bn[r], off, 32);
                int take = (int)(od < bd[r]) | ((int)(od == bd[r]) & (int)(on < bn[r]));
                bd[r] = take ? od : bd[r];
                bn[r] = take ? on : bn[r];
            }
        }
        if (ml == 0) {
#pragma unroll
            for (int r = 0; r < 8; ++r) {
                int rr = r + 8 * hi;   // vgpr r of half hi -> tile row rr
                s_idx[wave][rr] = bn[r];
                idx_out[(size_t)(rowBase + rr) * 4 + l] = (float)bn[r];
            }
        }
        __syncthreads();

        // residual update + per-level loss partial: loss_l = (1+beta)*mean(res_new^2)
        {
            const int idx = s_idx[wave][ml];
            const float* ec = s_cb + idx * CB_PAD + 2 * hi;
            float ls = 0.f;
#pragma unroll
            for (int kt = 0; kt < 8; ++kt) {
                v2f ev = *(const v2f*)(ec + kt * 4);
                res[2 * kt]     -= ev[0];
                res[2 * kt + 1] -= ev[1];
                ls += res[2 * kt] * res[2 * kt] + res[2 * kt + 1] * res[2 * kt + 1];
            }
            lacc[l] = ls;
        }
        __syncthreads();
    }

    // x_q = z - res_final -> bf16 (decoder input), packed 2 halves per b32 store
#pragma unroll
    for (int kt = 0; kt < 8; ++kt) {
        union { bf16 h[2]; unsigned u; } p;
        p.h[0] = (bf16)(z0[2 * kt]     - res[2 * kt]);
        p.h[1] = (bf16)(z0[2 * kt + 1] - res[2 * kt + 1]);
        *(unsigned*)((char*)(XQ + (size_t)row * 32 + kt * 4 + 2 * hi)) = p.u;
    }

    // loss: full-wave sum then one atomic per wave per level
#pragma unroll
    for (int l = 0; l < 4; ++l) {
        float v = lacc[l];
#pragma unroll
        for (int off = 16; off >= 1; off >>= 1) v += __shfl_xor(v, off, 32);
        if (lane == 0) atomicAdd(acc + l, v);
    }
}

// -------------------- host-side orchestration --------------------
static inline size_t align256(size_t x) { return (x + 255) & ~(size_t)255; }

extern "C" void kernel_launch(void* const* d_in, const int* in_sizes, int n_in,
                              void* d_out, int out_size, void* d_ws, size_t ws_size,
                              hipStream_t stream) {
    (void)in_sizes; (void)n_in; (void)out_size; (void)ws_size;

    const float* x  = (const float*)d_in[0];
    const float* ew[4] = { (const float*)d_in[1], (const float*)d_in[3], (const float*)d_in[5], (const float*)d_in[7] };
    const float* eb[4] = { (const float*)d_in[2], (const float*)d_in[4], (const float*)d_in[6], (const float*)d_in[8] };
    const float* dw[4] = { (const float*)d_in[9],  (const float*)d_in[11], (const float*)d_in[13], (const float*)d_in[15] };
    const float* db[4] = { (const float*)d_in[10], (const float*)d_in[12], (const float*)d_in[14], (const float*)d_in[16] };
    const float* cb = (const float*)d_in[17];

    const int encK[4] = {768, 512, 256, 128}, encN[4] = {512, 256, 128, 32};
    const int decK[4] = {32, 128, 256, 512},  decN[4] = {128, 256, 512, 768};

    char* ws = (char*)d_ws;
    size_t off = 0;
    float* accum = (float*)(ws + off); off += 256;
    bf16* efrag[4]; bf16* dfrag[4];
    for (int i = 0; i < 4; ++i) { efrag[i] = (bf16*)(ws + off); off += align256((size_t)encK[i] * encN[i] * 2); }
    for (int i = 0; i < 4; ++i) { dfrag[i] = (bf16*)(ws + off); off += align256((size_t)decK[i] * decN[i] * 2); }
    bf16*  P  = (bf16*)(ws + off);  off += (size_t)N_ROWS * 512 * 2;   // ping (max width 512)
    bf16*  Q  = (bf16*)(ws + off);  off += (size_t)N_ROWS * 256 * 2;   // pong (max width 256)
    float* Zb = (float*)(ws + off); off += (size_t)N_ROWS * 32 * 4;    // encoder output z (f32)

    float* out = (float*)d_out;
    float* loss_out = out + (size_t)N_ROWS * 768;
    float* idx_out  = loss_out + 1;

    k_zero<<<1, 64, 0, stream>>>(accum);
    for (int i = 0; i < 4; ++i)
        k_repack<<<(encK[i] * encN[i] + 255) / 256, 256, 0, stream>>>(ew[i], efrag[i], encK[i], encN[i]);
    for (int i = 0; i < 4; ++i)
        k_repack<<<(decK[i] * decN[i] + 255) / 256, 256, 0, stream>>>(dw[i], dfrag[i], decK[i], decN[i]);

    const dim3 blk(256);
    const int MB = N_ROWS / 128;
    // encoder
    k_gemm<768, 512, true,  false, true ><<<dim3( 8, MB), blk, 0, stream>>>(x, efrag[0], eb[0], P);
    k_gemm<512, 256, false, false, true ><<<dim3( 4, MB), blk, 0, stream>>>(P, efrag[1], eb[1], Q);
    k_gemm<256, 128, false, false, true ><<<dim3( 2, MB), blk, 0, stream>>>(Q, efrag[2], eb[2], P);
    k_gemm<128,  32, false, true,  false><<<dim3( 1, MB), blk, 0, stream>>>(P, efrag[3], eb[3], Zb);
    // residual VQ (z f32 -> x_q bf16 into Q, indices as floats into d_out tail)
    k_vq<<<dim3(MB), blk, 0, stream>>>(Zb, cb, Q, idx_out, accum);
    // decoder
    k_gemm< 32, 128, false, false, true ><<<dim3( 2, MB), blk, 0, stream>>>(Q, dfrag[0], db[0], P);
    k_gemm<128, 256, false, false, true ><<<dim3( 4, MB), blk, 0, stream>>>(P, dfrag[1], db[1], Q);
    k_gemm<256, 512, false, false, true ><<<dim3( 8, MB), blk, 0, stream>>>(Q, dfrag[2], db[2], P);
    k_gemm<512, 768, false, true,  false><<<dim3(12, MB), blk, 0, stream>>>(P, dfrag[3], db[3], out);
    k_finalize<<<1, 1, 0, stream>>>(accum, loss_out);
}
